// Decoder_15934328668408
// MI455X (gfx1250) — matrix operand order repaired and checked by reading, not compiled, not run on hardware
//
#include <hip/hip_runtime.h>
#include <hip/hip_bf16.h>

// ---------------- CDNA5 types ----------------
typedef __bf16 bf16;
typedef __attribute__((ext_vector_type(16))) __bf16 v16bf;
typedef __attribute__((ext_vector_type(8)))  __bf16 v8bf;
typedef __attribute__((ext_vector_type(8)))  float  v8f;

#define B_  256
#define T_  256
#define M_  256
#define P_  256
#define ROWS 16           // batch rows per workgroup == WMMA M
#define DEC_THREADS 512   // 16 waves (wave32)

// ---------------- workspace layout (bytes) ----------------
// ehbf : B*T*M bf16          = 33,554,432
// Wrt  : (4P x P)  bf16 N-major = 524,288
// Wwt  : (M x 2P)  bf16 N-major = 262,144
// Wut  : (M x M)   bf16 N-major = 131,072
// Ue   : B*T*M f32           = 67,108,864
#define WS_EHBF 0
#define WS_WRT  33554432
#define WS_WWT  34078720
#define WS_WUT  34340864
#define WS_UE   34471936

__device__ __forceinline__ float sigmoid_(float x) {
    return 1.0f / (1.0f + __expf(-x));
}
__device__ __forceinline__ float tanh_(float x) {
    // 1 - 2/(exp(2x)+1): single v_exp_f32, saturates correctly for large |x|
    float e = __expf(2.0f * x);
    return 1.0f - 2.0f / (e + 1.0f);
}

// Build a 16x32 bf16 A-fragment: lanes 0-15 hold K=[k0..k0+7],[k0+16..k0+23],
// lanes 16-31 hold K=[k0+8..k0+15],[k0+24..k0+31] (ISA 7.12.2 layout).
__device__ __forceinline__ v16bf load_a_frag(const bf16* rowp, int k0, int half) {
    const v8bf lo = *(const v8bf*)(rowp + k0 + half * 8);
    const v8bf hi = *(const v8bf*)(rowp + k0 + half * 8 + 16);
    return __builtin_shufflevector(lo, hi, 0,1,2,3,4,5,6,7,8,9,10,11,12,13,14,15);
}
// B-fragment: same K layout as A-fragment for its column n (weights stored N-major).
__device__ __forceinline__ v16bf load_b_frag(const bf16* colp, int k0, int half) {
    const v8bf lo = *(const v8bf*)(colp + k0 + half * 8);
    const v8bf hi = *(const v8bf*)(colp + k0 + half * 8 + 16);
    return __builtin_shufflevector(lo, hi, 0,1,2,3,4,5,6,7,8,9,10,11,12,13,14,15);
}

// ---------------- kernel 1: f32 -> bf16 convert (+ weight transpose) ----------------
__global__ void convert_kernel(const float* __restrict__ eh,
                               const float* __restrict__ Wr,
                               const float* __restrict__ Ww,
                               const float* __restrict__ Wu,
                               bf16* __restrict__ ehbf,
                               bf16* __restrict__ Wrt,
                               bf16* __restrict__ Wwt,
                               bf16* __restrict__ Wut) {
    const long N_EH = (long)B_ * T_ * M_;         // 16,777,216
    const long N_WR = 4L * P_ * P_;               // 1,048,576? no: 1024*256 = 262,144
    const long NWR  = (long)(4 * P_) * P_;        // 262,144
    const long NWW  = (long)M_ * (2 * P_);        // 131,072
    const long NWU  = (long)M_ * M_;              // 65,536
    (void)N_WR;
    const long total = N_EH + NWR + NWW + NWU;
    for (long i = (long)blockIdx.x * blockDim.x + threadIdx.x; i < total;
         i += (long)gridDim.x * blockDim.x) {
        if (i < N_EH) {
            ehbf[i] = (bf16)eh[i];
        } else if (i < N_EH + NWR) {
            long j = i - N_EH;               // j = n*256 + k, n in [0,1024)
            int n = (int)(j >> 8), k = (int)(j & 255);
            Wrt[j] = (bf16)Wr[(size_t)k * (4 * P_) + n];
        } else if (i < N_EH + NWR + NWW) {
            long j = i - N_EH - NWR;         // j = n*512 + k, n in [0,256)
            int n = (int)(j >> 9), k = (int)(j & 511);
            Wwt[j] = (bf16)Ww[(size_t)k * M_ + n];
        } else {
            long j = i - N_EH - NWR - NWW;   // j = n*256 + k
            int n = (int)(j >> 8), k = (int)(j & 255);
            Wut[j] = (bf16)Wu[(size_t)k * M_ + n];
        }
    }
}

// ---------------- kernel 2: Ue = encoder_h @ Wu + bu  (WMMA bf16) ----------------
// One 16x16 output tile per wave. Rows = B*T = 65536 -> 4096 row tiles, 16 col tiles.
__global__ void ue_wmma_kernel(const bf16* __restrict__ ehbf,
                               const bf16* __restrict__ Wut,
                               const float* __restrict__ bu,
                               float* __restrict__ Ue) {
    const int tid  = threadIdx.x;
    const int lane = tid & 31;
    const int gw   = blockIdx.x * (blockDim.x >> 5) + (tid >> 5);
    const int tileRow = gw >> 4;     // 0..4095
    const int tileCol = gw & 15;     // 0..15
    const int half = lane >> 4;
    const int nl   = lane & 15;
    const int col  = tileCol * 16 + nl;

    const bf16* arow = ehbf + (size_t)(tileRow * 16 + nl) * M_;
    const bf16* bcol = Wut + (size_t)col * M_;

    v8f c = {};
#pragma unroll
    for (int kk = 0; kk < 8; ++kk) {
        const int k0 = kk * 32;
        v16bf a = load_a_frag(arow, k0, half);
        v16bf b = load_b_frag(bcol, k0, half);
        c = __builtin_amdgcn_wmma_f32_16x16x32_bf16(false, a, false, b,
                                                    (short)0, c, false, false);
    }
    const float buv = bu[col];
    const int rbase = half * 8;
#pragma unroll
    for (int r = 0; r < 8; ++r) {
        const int row = tileRow * 16 + rbase + r;
        Ue[(size_t)row * M_ + col] = c[r] + buv;
    }
}

// ---------------- kernel 3: persistent decoder scan ----------------
// 16 workgroups x 512 threads; workgroup g owns batch rows [16g, 16g+16).
// Entire 255-step recurrence runs inside the workgroup (only s_barrier syncs).
__global__ void decoder_kernel(const float* __restrict__ xdata,     // (B, T-1)
                               const float* __restrict__ encoder_h, // (B, T, M) f32
                               const float* __restrict__ h0,
                               const float* __restrict__ s0,
                               const float* __restrict__ Wd,        // (1+M)
                               const float* __restrict__ bd,
                               const float* __restrict__ Wk,        // (4P)
                               const bf16*  __restrict__ Wrt,       // (4P x P) N-major
                               const float* __restrict__ bl,        // (4P)
                               const bf16*  __restrict__ Wwt,       // (M x 2P) N-major
                               const float* __restrict__ bw,        // (M)
                               const float* __restrict__ Wv,        // (M)
                               const float* __restrict__ bv,
                               const float* __restrict__ Wvb,       // (P+M)
                               const float* __restrict__ bvb,
                               const float* __restrict__ Wwb,       // (P)
                               const float* __restrict__ bwb,       // (P)
                               const float* __restrict__ Ue,        // (B,T,M) f32
                               float* __restrict__ out) {           // (B,1,P)
    extern __shared__ char smem_raw[];
    float* s_s   = (float*)smem_raw;        // [16][256] cell state (f32)
    float* s_ctx = s_s   + ROWS * P_;       // [16][256] context
    float* s_dsc = s_ctx + ROWS * M_;       // [16][256] attention proj
    float* s_l   = s_dsc + ROWS * M_;       // [16][256] logits -> beta
    float* s_wk  = s_l   + ROWS * T_;       // [1024]
    float* s_bl  = s_wk  + 4 * P_;          // [1024]
    float* s_wv  = s_bl  + 4 * P_;          // [256]
    float* s_wd  = s_wv  + M_;              // [260]
    float* s_y   = s_wd  + 260;             // [16]
    bf16*  s_hs  = (bf16*)(s_y + 16);       // [16][512] = [h | s] bf16 WMMA A-panel

    const int tid  = threadIdx.x;
    const int wv   = tid >> 5;              // wave 0..15
    const int lane = tid & 31;
    const int half = lane >> 4;
    const int nl   = lane & 15;
    const int b0   = blockIdx.x * ROWS;

    const float bd0  = bd[0];
    const float bv0  = bv[0];
    const float bvb0 = bvb[0];

    // ---- init: cache small weights + load initial state ----
    for (int i = tid; i < 4 * P_; i += DEC_THREADS) { s_wk[i] = Wk[i]; s_bl[i] = bl[i]; }
    for (int i = tid; i < 1 + M_; i += DEC_THREADS) s_wd[i] = Wd[i];
    if (tid < M_) s_wv[tid] = Wv[tid];
    for (int idx = tid; idx < ROWS * P_; idx += DEC_THREADS) {
        const int r = idx >> 8, m = idx & 255;
        const float hv = h0[(size_t)(b0 + r) * P_ + m];
        const float sv = s0[(size_t)(b0 + r) * P_ + m];
        s_s[idx]   = sv;
        s_ctx[idx] = 0.0f;
        s_hs[r * 512 + m]       = (bf16)hv;
        s_hs[r * 512 + 256 + m] = (bf16)sv;
    }
    __syncthreads();

    for (int t = 0; t < T_ - 1; ++t) {
        // ---- phase A: y = [x_t, ctx] @ Wd + bd  (wave wv owns row wv) ----
        {
            float acc = 0.0f;
#pragma unroll
            for (int q = 0; q < 8; ++q) {
                const int m = lane + 32 * q;
                acc += s_ctx[wv * M_ + m] * s_wd[1 + m];
            }
#pragma unroll
            for (int off = 16; off > 0; off >>= 1) acc += __shfl_xor(acc, off, 32);
            if (lane == 0) {
                const float xv = xdata[(size_t)(b0 + wv) * (T_ - 1) + t];
                s_y[wv] = xv * s_wd[0] + acc + bd0;
            }
        }
        __syncthreads();

        // ---- phase B: z = y*Wk + h@Wr + bl ; gates -> h_new, s_new ----
        // Wave wv owns unit block [wv*16, wv*16+16) for all four gates (i,f,g,o).
        v8f ci = {}, cf = {}, cg = {}, co = {};
        {
            const int nb = wv * 16 + nl;
            const bf16* bi = Wrt + (size_t)(nb)             * P_;
            const bf16* bf = Wrt + (size_t)(P_ + nb)        * P_;
            const bf16* bg = Wrt + (size_t)(2 * P_ + nb)    * P_;
            const bf16* bo = Wrt + (size_t)(3 * P_ + nb)    * P_;
            const bf16* arow = s_hs + nl * 512;             // K = h part (0..255)
#pragma unroll
            for (int kk = 0; kk < 8; ++kk) {
                const int k0 = kk * 32;
                v16bf a = load_a_frag(arow, k0, half);
                ci = __builtin_amdgcn_wmma_f32_16x16x32_bf16(false, a, false,
                        load_b_frag(bi, k0, half), (short)0, ci, false, false);
                cf = __builtin_amdgcn_wmma_f32_16x16x32_bf16(false, a, false,
                        load_b_frag(bf, k0, half), (short)0, cf, false, false);
                cg = __builtin_amdgcn_wmma_f32_16x16x32_bf16(false, a, false,
                        load_b_frag(bg, k0, half), (short)0, cg, false, false);
                co = __builtin_amdgcn_wmma_f32_16x16x32_bf16(false, a, false,
                        load_b_frag(bo, k0, half), (short)0, co, false, false);
            }
        }
        __syncthreads();   // all waves finished reading old h panel
        {
            const int u = wv * 16 + nl;
            const int rbase = half * 8;
#pragma unroll
            for (int r = 0; r < 8; ++r) {
                const int row = rbase + r;
                const float yv = s_y[row];
                const float iv = ci[r] + yv * s_wk[u]            + s_bl[u];
                const float fv = cf[r] + yv * s_wk[P_ + u]       + s_bl[P_ + u];
                const float gv = cg[r] + yv * s_wk[2 * P_ + u]   + s_bl[2 * P_ + u];
                const float ov = co[r] + yv * s_wk[3 * P_ + u]   + s_bl[3 * P_ + u];
                const float sold = s_s[row * P_ + u];
                const float sn = sigmoid_(fv) * sold + sigmoid_(iv) * tanh_(gv);
                const float hn = sigmoid_(ov) * tanh_(sn);
                s_s[row * P_ + u] = sn;
                s_hs[row * 512 + u]        = (bf16)hn;
                s_hs[row * 512 + 256 + u]  = (bf16)sn;
            }
        }
        __syncthreads();

        // ---- phase C: dsc = [h_new, s_new] @ Ww + bw  (K=512; wave wv -> N tile wv) ----
        {
            v8f cd = {};
            const int nb = wv * 16 + nl;
            const bf16* bcol = Wwt + (size_t)nb * (2 * P_);
            const bf16* arow = s_hs + nl * 512;
#pragma unroll
            for (int kk = 0; kk < 16; ++kk) {
                const int k0 = kk * 32;
                v16bf a = load_a_frag(arow, k0, half);
                v16bf b = load_b_frag(bcol, k0, half);
                cd = __builtin_amdgcn_wmma_f32_16x16x32_bf16(false, a, false, b,
                                                             (short)0, cd, false, false);
            }
            const float bwv = bw[nb];
            const int rbase = half * 8;
#pragma unroll
            for (int r = 0; r < 8; ++r) s_dsc[(rbase + r) * M_ + nb] = cd[r] + bwv;
        }
        __syncthreads();

        // ---- phase D: l[b,t] = tanh(dsc[b]+Ue[b,t]) . Wv + bv  (L2-resident Ue) ----
        {
            const int ta = tid & 255;
            const int jb = (tid >> 8) * 8;
            const float4* w4 = (const float4*)s_wv;
#pragma unroll 2
            for (int j = 0; j < 8; ++j) {
                const int br = jb + j;
                const float4* ue4 = (const float4*)(Ue + ((size_t)(b0 + br) * T_ + ta) * M_);
                const float4* d4  = (const float4*)(s_dsc + br * M_);
                float acc = 0.0f;
                for (int mq = 0; mq < M_ / 4; ++mq) {
                    const float4 u4 = ue4[mq];
                    const float4 dd = d4[mq];
                    const float4 wq = w4[mq];
                    acc += tanh_(dd.x + u4.x) * wq.x;
                    acc += tanh_(dd.y + u4.y) * wq.y;
                    acc += tanh_(dd.z + u4.z) * wq.z;
                    acc += tanh_(dd.w + u4.w) * wq.w;
                }
                s_l[br * T_ + ta] = acc + bv0;
            }
        }
        __syncthreads();

        // ---- phase E: softmax over t (wave wv owns row wv) ----
        {
            float vals[8];
            float mx = -3.0e38f;
#pragma unroll
            for (int q = 0; q < 8; ++q) {
                vals[q] = s_l[wv * T_ + lane + 32 * q];
                mx = fmaxf(mx, vals[q]);
            }
#pragma unroll
            for (int off = 16; off > 0; off >>= 1) mx = fmaxf(mx, __shfl_xor(mx, off, 32));
            float sum = 0.0f;
#pragma unroll
            for (int q = 0; q < 8; ++q) { vals[q] = __expf(vals[q] - mx); sum += vals[q]; }
#pragma unroll
            for (int off = 16; off > 0; off >>= 1) sum += __shfl_xor(sum, off, 32);
            const float inv = 1.0f / sum;
#pragma unroll
            for (int q = 0; q < 8; ++q) s_l[wv * T_ + lane + 32 * q] = vals[q] * inv;
        }
        __syncthreads();

        // ---- phase F: ctx[b,:] = beta[b,:] @ encoder_h[b]  (coalesced over m) ----
        {
            const int m = tid & 255;
            const int jb = (tid >> 8) * 8;
#pragma unroll 2
            for (int j = 0; j < 8; ++j) {
                const int br = jb + j;
                const float* eh = encoder_h + (size_t)(b0 + br) * T_ * M_ + m;
                const float* bt = s_l + br * T_;
                float acc = 0.0f;
                for (int tt = 0; tt < T_; tt += 4) {
                    acc += bt[tt]     * eh[(size_t)tt * M_];
                    acc += bt[tt + 1] * eh[(size_t)(tt + 1) * M_];
                    acc += bt[tt + 2] * eh[(size_t)(tt + 2) * M_];
                    acc += bt[tt + 3] * eh[(size_t)(tt + 3) * M_];
                }
                s_ctx[br * M_ + m] = acc;
            }
        }
        __syncthreads();
    }

    // ---- output head: out[b,:] = (([h,ctx].Wvb)+bvb) * Wwb + bwb (wave wv -> row wv) ----
    {
        float acc = 0.0f;
#pragma unroll
        for (int q = 0; q < 16; ++q) {
            const int k2 = lane + 32 * q;  // 0..511
            const float catv = (k2 < P_) ? (float)s_hs[wv * 512 + k2]
                                         : s_ctx[wv * M_ + (k2 - P_)];
            acc += catv * Wvb[k2];
        }
#pragma unroll
        for (int off = 16; off > 0; off >>= 1) acc += __shfl_xor(acc, off, 32);
        const float vfin = acc + bvb0;
#pragma unroll
        for (int q = 0; q < 8; ++q) {
            const int p = lane + 32 * q;
            out[(size_t)(b0 + wv) * P_ + p] = vfin * Wwb[p] + bwb[p];
        }
    }
}

// ---------------- host launcher ----------------
extern "C" void kernel_launch(void* const* d_in, const int* in_sizes, int n_in,
                              void* d_out, int out_size, void* d_ws, size_t ws_size,
                              hipStream_t stream) {
    (void)in_sizes; (void)n_in; (void)out_size; (void)ws_size;
    const float* data      = (const float*)d_in[0];
    const float* encoder_h = (const float*)d_in[1];
    const float* h0        = (const float*)d_in[2];
    const float* s0        = (const float*)d_in[3];
    const float* Wd        = (const float*)d_in[4];
    const float* bd        = (const float*)d_in[5];
    const float* Wk        = (const float*)d_in[6];
    const float* Wr        = (const float*)d_in[7];
    const float* bl        = (const float*)d_in[8];
    const float* Ww        = (const float*)d_in[9];
    const float* bw        = (const float*)d_in[10];
    const float* Wu        = (const float*)d_in[11];
    const float* bu        = (const float*)d_in[12];
    const float* Wv        = (const float*)d_in[13];
    const float* bv        = (const float*)d_in[14];
    const float* Wvb       = (const float*)d_in[15];
    const float* bvb       = (const float*)d_in[16];
    const float* Wwb       = (const float*)d_in[17];
    const float* bwb       = (const float*)d_in[18];
    float* out = (float*)d_out;

    char* ws = (char*)d_ws;
    bf16*  ehbf = (bf16*)(ws + WS_EHBF);
    bf16*  Wrt  = (bf16*)(ws + WS_WRT);
    bf16*  Wwt  = (bf16*)(ws + WS_WWT);
    bf16*  Wut  = (bf16*)(ws + WS_WUT);
    float* Ue   = (float*)(ws + WS_UE);

    // 1) bf16 convert + transpose of WMMA operands
    convert_kernel<<<8192, 256, 0, stream>>>(encoder_h, Wr, Ww, Wu, ehbf, Wrt, Wwt, Wut);

    // 2) Ue = encoder_h @ Wu + bu : 65536 WMMA tiles, 8 waves/block -> 8192 blocks
    ue_wmma_kernel<<<8192, 256, 0, stream>>>(ehbf, Wut, bu, Ue);

    // 3) persistent decoder: one workgroup per 16-row batch tile
    constexpr size_t SM_FLOATS =
        (size_t)ROWS * P_ /*s*/ + ROWS * M_ /*ctx*/ + ROWS * M_ /*dsc*/ + ROWS * T_ /*l*/
        + 4 * P_ /*wk*/ + 4 * P_ /*bl*/ + M_ /*wv*/ + 260 /*wd*/ + 16 /*y*/;
    constexpr size_t SMEM_BYTES = SM_FLOATS * sizeof(float) + (size_t)ROWS * 512 * sizeof(bf16);
    decoder_kernel<<<B_ / ROWS, DEC_THREADS, SMEM_BYTES, stream>>>(
        data, encoder_h, h0, s0, Wd, bd, Wk, Wrt, bl, Wwt, bw,
        Wv, bv, Wvb, bvb, Wwb, bwb, Ue, out);
}
